// SegmentedAttention_22342419873961
// MI455X (gfx1250) — matrix operand order repaired and checked by reading, not compiled, not run on hardware
//
#include <hip/hip_runtime.h>
#include <hip/hip_bf16.h>

// ---------------------------------------------------------------------------
// Problem constants (match reference)
// ---------------------------------------------------------------------------
#define B_    2
#define S_    2048
#define D_    2048
#define H_    16
#define DH_   128
#define P_    16
#define HD_   2048            // H*DH
#define QKVN_ 6144            // 3*H*DH
#define SP_   2064            // S + P   (logical keys)
#define SPP_  2080            // padded key stride (multiple of 32)
#define NTOK_ 4096            // B*S
#define EPSLN 1e-5f
#define QSCALE 0.08838834764831845f   // DH^-0.5

typedef __attribute__((ext_vector_type(16))) _Float16 v16h;
typedef __attribute__((ext_vector_type(8)))  _Float16 v8h;
typedef __attribute__((ext_vector_type(8)))  float    v8f;
typedef __attribute__((ext_vector_type(4)))  unsigned int v4u;
typedef __attribute__((ext_vector_type(8)))  int      v8i_t;
typedef __attribute__((ext_vector_type(4)))  int      v4i_t;

// ---------------------------------------------------------------------------
// WMMA helpers (CDNA5 16x16x32 f16 -> f32 accumulate)
// ---------------------------------------------------------------------------
__device__ __forceinline__ v8f wmma_f16(v16h a, v16h b, v8f c) {
  return __builtin_amdgcn_wmma_f32_16x16x32_f16(
      /*neg_a=*/false, a, /*neg_b=*/false, b,
      /*c_mod=*/(short)0, c, /*reuse_a=*/false, /*reuse_b=*/false);
}

// A-matrix fragment (16x32, 16-bit): lane L holds row M=L%16;
// lanes 0-15 hold K={0..7,16..23}, lanes 16-31 hold K={8..15,24..31}.
__device__ __forceinline__ v16h load_a(const _Float16* base, int ldm) {
  const int lane = threadIdx.x & 31;
  const _Float16* p = base + (size_t)(lane & 15) * ldm + ((lane >> 4) << 3);
  v8h lo = *(const v8h*)(p);
  v8h hi = *(const v8h*)(p + 16);
  v16h o;
#pragma unroll
  for (int i = 0; i < 8; ++i) { o[i] = lo[i]; o[i + 8] = hi[i]; }
  return o;
}

// B-matrix fragment (32x16, 16-bit): lane L holds column N=L%16;
// lanes 0-15 hold K={0..7,16..23}, lanes 16-31 hold K={8..15,24..31}.
// Memory is stored N-major: element (k,n) at base + n*ldm + k.
__device__ __forceinline__ v16h load_b(const _Float16* base, int ldm) {
  const int lane = threadIdx.x & 31;
  const _Float16* p = base + (size_t)(lane & 15) * ldm + ((lane >> 4) << 3);
  v8h lo = *(const v8h*)(p);
  v8h hi = *(const v8h*)(p + 16);
  v16h o;
#pragma unroll
  for (int i = 0; i < 8; ++i) { o[i] = lo[i]; o[i + 8] = hi[i]; }
  return o;
}

// ---------------------------------------------------------------------------
// Tensor Data Mover: async 2-D tile load (rows x dim0 f16 elements) into LDS.
// D# layout per CDNA5 ISA 8.3/8.4: group0 = {flags, lds_addr, global_addr,
// type=2}; group1 = {data_size=2B, tensor/tile dims, dim0 stride}.
// This toolchain exposes the 6-arg builtin:
//   (uint32x4 g0, int32x8 g1, int32x4, int32x4, int32x8, i32 cpol)
// ---------------------------------------------------------------------------
__device__ __forceinline__ void tdm_load_2d(unsigned lds_addr, const void* gptr,
                                            unsigned dim0_elems, unsigned rows,
                                            unsigned stride0_elems) {
  const unsigned long long ga = (unsigned long long)gptr;
  v4u g0;
  g0[0] = 1u;                                          // count=1, user D#
  g0[1] = lds_addr;                                    // LDS byte address
  g0[2] = (unsigned)(ga & 0xFFFFFFFFu);                // global_addr[31:0]
  g0[3] = (unsigned)((ga >> 32) & 0x01FFFFFFu) | (2u << 30);  // addr[56:32]|type=2
  v8i_t g1;
  g1[0] = (int)(1u << 16);                             // data_size=1 -> 2 bytes
  g1[1] = (int)((dim0_elems & 0xFFFFu) << 16);         // tensor_dim0[15:0]
  g1[2] = (int)((dim0_elems >> 16) | ((rows & 0xFFFFu) << 16));  // d0 hi|d1 lo
  g1[3] = (int)((rows >> 16) | ((dim0_elems & 0xFFFFu) << 16));  // d1 hi|tile_dim0
  g1[4] = (int)(rows & 0xFFFFu);                       // tile_dim1 (tile_dim2=0)
  g1[5] = (int)stride0_elems;                          // tensor_dim0_stride[31:0]
  g1[6] = 0;                                           // stride0 hi | stride1 lo
  g1[7] = 0;
  v4i_t z4 = {};
  v8i_t z8 = {};
  __builtin_amdgcn_tensor_load_to_lds(g0, g1, z4, z4, z8, 0);
}

// ---------------------------------------------------------------------------
// Workgroup-cooperative GEMM core: 8 waves, C tile = 32(M) x 512(N).
// TDM double-buffers the 32xKC A panel in LDS; each wave owns 64 N columns
// (2 A frags x 4 B frags = 8 WMMAs per 32-K step, B reused across A frags).
// ---------------------------------------------------------------------------
template <int KTOT>
__device__ __forceinline__ void gemm32x64_core(
    const _Float16* __restrict__ Ag,   // global A, rows m0.., row stride KTOT
    const _Float16* __restrict__ Bn,   // wT + n0*KTOT (N-major, row stride KTOT)
    _Float16* lds,                     // 2 * 32 * 512 f16 (64 KB)
    v8f (&acc)[2][4]) {
  constexpr int KC = 512;
  constexpr int NC = KTOT / KC;
  const int wslot = threadIdx.x >> 5;

  if (wslot == 0)
    tdm_load_2d((unsigned)(size_t)lds, Ag, KC, 32, KTOT);

#pragma unroll
  for (int c = 0; c < NC; ++c) {
    _Float16* buf = lds + (c & 1) * (32 * KC);
    if (wslot == 0) {
      if (c + 1 < NC) {
        tdm_load_2d((unsigned)(size_t)(lds + ((c + 1) & 1) * (32 * KC)),
                    Ag + (size_t)(c + 1) * KC, KC, 32, KTOT);
        __builtin_amdgcn_s_wait_tensorcnt(1);   // chunk c landed, c+1 in flight
      } else {
        __builtin_amdgcn_s_wait_tensorcnt(0);   // final chunk landed
      }
    }
    __syncthreads();

    const _Float16* bcol = Bn + (size_t)c * KC;
    for (int k0 = 0; k0 < KC; k0 += 32) {
      if (k0 + 256 < KC) __builtin_prefetch(bcol + k0 + 256, 0, 1);
      const v16h a0 = load_a(buf + k0, KC);
      const v16h a1 = load_a(buf + 16 * KC + k0, KC);
#pragma unroll
      for (int j = 0; j < 4; ++j) {
        const v16h b = load_b(bcol + (size_t)(j * 16) * KTOT + k0, KTOT);
        acc[0][j] = wmma_f16(a0, b, acc[0][j]);
        acc[1][j] = wmma_f16(a1, b, acc[1][j]);
      }
    }
    __asm__ volatile("s_wait_dscnt 0" ::: "memory");  // drain LDS reads
    __syncthreads();                                  // before TDM overwrites
  }
}

// ---------------------------------------------------------------------------
// Kernel 1: LayerNorm -> f16  (one block per token)
// ---------------------------------------------------------------------------
__global__ void ln_kernel(const float* __restrict__ x,
                          const float* __restrict__ g,
                          const float* __restrict__ bln,
                          _Float16* __restrict__ xn) {
  const int tok = blockIdx.x;
  const int tid = threadIdx.x;
  const float* xr = x + (size_t)tok * D_;
  float v[8];
  float s = 0.f, s2 = 0.f;
#pragma unroll
  for (int i = 0; i < 8; ++i) {
    v[i] = xr[tid * 8 + i];
    s += v[i];
    s2 += v[i] * v[i];
  }
#pragma unroll
  for (int off = 1; off < 32; off <<= 1) {
    s  += __shfl_xor(s,  off, 32);
    s2 += __shfl_xor(s2, off, 32);
  }
  __shared__ float rs[8], rs2[8];
  const int w = tid >> 5;
  if ((tid & 31) == 0) { rs[w] = s; rs2[w] = s2; }
  __syncthreads();
  if (w == 0) {
    float a  = (tid < 8) ? rs[tid]  : 0.f;
    float b2 = (tid < 8) ? rs2[tid] : 0.f;
#pragma unroll
    for (int off = 1; off < 8; off <<= 1) {
      a  += __shfl_xor(a,  off, 32);
      b2 += __shfl_xor(b2, off, 32);
    }
    if (tid == 0) { rs[0] = a; rs2[0] = b2; }
  }
  __syncthreads();
  const float mu   = rs[0] * (1.f / D_);
  const float var  = rs2[0] * (1.f / D_) - mu * mu;
  const float rstd = rsqrtf(var + EPSLN);
  _Float16* o = xn + (size_t)tok * D_;
#pragma unroll
  for (int i = 0; i < 8; ++i) {
    const int c = tid * 8 + i;
    o[c] = (_Float16)(((v[i] - mu) * rstd) * g[c] + bln[c]);
  }
}

// ---------------------------------------------------------------------------
// Kernel 2: per-(token, head) sigmoid mix gate (one wave each)
// ---------------------------------------------------------------------------
__global__ void mix_kernel(const _Float16* __restrict__ xn,
                           const float* __restrict__ w_mix,
                           const float* __restrict__ b_mix,
                           float* __restrict__ mix) {
  const int wid  = (int)((blockIdx.x * blockDim.x + threadIdx.x) >> 5);
  const int lane = threadIdx.x & 31;
  const int tok  = wid / H_;
  const int h    = wid - tok * H_;
  const _Float16* xr = xn + (size_t)tok * D_;
  float acc = 0.f;
#pragma unroll 4
  for (int i = 0; i < D_ / 32; ++i) {
    const int d = lane + i * 32;
    acc += (float)xr[d] * w_mix[(size_t)d * H_ + h];
  }
#pragma unroll
  for (int off = 1; off < 32; off <<= 1) acc += __shfl_xor(acc, off, 32);
  if (lane == 0)
    mix[(size_t)tok * H_ + h] = 1.f / (1.f + __expf(-(acc + b_mix[h])));
}

// ---------------------------------------------------------------------------
// Kernel 3: fp32 [K,N] -> f16 transposed [N,K] (weight prep for WMMA B-frags)
// ---------------------------------------------------------------------------
__global__ void tconv_kernel(const float* __restrict__ src,
                             _Float16* __restrict__ dst, int K, int N) {
  const size_t total = (size_t)K * N;
  for (size_t i = blockIdx.x * (size_t)blockDim.x + threadIdx.x; i < total;
       i += (size_t)gridDim.x * blockDim.x) {
    const size_t k = i / N;
    const size_t n = i - k * N;
    dst[n * K + k] = (_Float16)src[i];
  }
}

// ---------------------------------------------------------------------------
// Kernel 4: persistent-memory tokens into k rows [0,P) and vT cols [0,P)
// ---------------------------------------------------------------------------
__global__ void pmem_kernel(const float* __restrict__ pmem,
                            _Float16* __restrict__ k_h,
                            _Float16* __restrict__ v_t) {
  const int idx = blockIdx.x * blockDim.x + threadIdx.x;  // H*P*DH
  if (idx >= H_ * P_ * DH_) return;
  const int h   = idx / (P_ * DH_);
  const int rem = idx - h * (P_ * DH_);
  const int p   = rem / DH_;
  const int dh  = rem - p * DH_;
  const _Float16 kk = (_Float16)pmem[idx];
  const _Float16 vv = (_Float16)pmem[H_ * P_ * DH_ + idx];
#pragma unroll
  for (int b = 0; b < B_; ++b) {
    const size_t bh = (size_t)(b * H_ + h);
    k_h[(bh * SPP_ + p) * DH_ + dh]   = kk;
    v_t[(bh * DH_ + dh) * SPP_ + p]   = vv;
  }
}

// Zero the padded key tail [SP_, SPP_) so masked WMMA inputs are finite.
__global__ void pad_kernel(_Float16* __restrict__ k_h,
                           _Float16* __restrict__ v_t) {
  const int idx = blockIdx.x * blockDim.x + threadIdx.x;
  if (idx >= B_ * H_ * (SPP_ - SP_) * DH_) return;
  const int bh  = idx / ((SPP_ - SP_) * DH_);
  const int rem = idx - bh * ((SPP_ - SP_) * DH_);
  const int pk  = rem / DH_;
  const int dh  = rem - pk * DH_;
  k_h[((size_t)bh * SPP_ + SP_ + pk) * DH_ + dh] = (_Float16)0.f;
  v_t[((size_t)bh * DH_ + dh) * SPP_ + SP_ + pk] = (_Float16)0.f;
}

// ---------------------------------------------------------------------------
// Kernel 5: QKV GEMM (TDM-staged A, f16 WMMA) + fused split/scale/mix epilogue.
// Workgroup = 8 waves = 32x512 C tile of xn[4096,2048] @ w_qkv[2048,6144].
// ---------------------------------------------------------------------------
__global__ void qkv_gemm_kernel(const _Float16* __restrict__ xn,
                                const _Float16* __restrict__ wT,   // [QKVN_, D_]
                                const float* __restrict__ mix,
                                const float* __restrict__ vres,    // [B,H,S,DH]
                                _Float16* __restrict__ q_h,
                                _Float16* __restrict__ k_h,
                                _Float16* __restrict__ v_t) {
  __shared__ _Float16 lds_a[2 * 32 * 512];   // 64 KB double-buffered A panel
  const int lane  = threadIdx.x & 31;
  const int wslot = threadIdx.x >> 5;
  const int cb = blockIdx.x % (QKVN_ / 512);
  const int rb = blockIdx.x / (QKVN_ / 512);
  const int m0 = rb * 32;
  const int n0 = cb * 512 + wslot * 64;

  v8f acc[2][4] = {};
  gemm32x64_core<D_>(xn + (size_t)m0 * D_, wT + (size_t)n0 * D_, lds_a, acc);

  const int rbase = (lane >> 4) << 3;   // 0 or 8
  const int nl    = lane & 15;
#pragma unroll
  for (int t = 0; t < 2; ++t) {
#pragma unroll
    for (int j = 0; j < 4; ++j) {
      const int c   = n0 + j * 16 + nl;
      const int sec = c >> 11;            // 0:q 1:k 2:v
      const int ch  = c & 2047;
      const int h   = ch >> 7;
      const int dh  = ch & 127;
#pragma unroll
      for (int r = 0; r < 8; ++r) {
        const int token = m0 + t * 16 + r + rbase;
        const int b = token >> 11;
        const int s = token & (S_ - 1);
        const size_t bh = (size_t)(b * H_ + h);
        const float val = acc[t][j][r];
        if (sec == 0) {
          q_h[(bh * S_ + s) * DH_ + dh] = (_Float16)(val * QSCALE);
        } else if (sec == 1) {
          k_h[(bh * SPP_ + P_ + s) * DH_ + dh] = (_Float16)val;
        } else {
          const float mx = mix[(size_t)token * H_ + h];
          const float vr = vres[(bh * S_ + s) * DH_ + dh];
          v_t[(bh * DH_ + dh) * SPP_ + P_ + s] =
              (_Float16)(val * (1.f - mx) + vr * mx);
        }
      }
    }
  }
}

// ---------------------------------------------------------------------------
// Kernel 6: flash attention. One wave per 16-query tile of one (b,h).
// 32-key blocks: 8 score WMMAs + online softmax + 8 PV WMMAs.
// ---------------------------------------------------------------------------
__global__ void attn_kernel(const _Float16* __restrict__ q_h,
                            const _Float16* __restrict__ k_h,
                            const _Float16* __restrict__ v_t,
                            _Float16* __restrict__ ao) {
  __shared__ _Float16 lds_p[8][16 * 32];   // per-wave 16x32 prob tile
  const int wslot = threadIdx.x >> 5;
  const int lane  = threadIdx.x & 31;
  const int wid   = blockIdx.x * 8 + wslot;
  const int mt = wid & (S_ / 16 - 1);      // 128 query tiles per (b,h)
  const int bh = wid >> 7;                 // 0..31
  const int m0 = mt * 16;

  const _Float16* qbase = q_h + ((size_t)bh * S_ + m0) * DH_;
  const _Float16* kbase = k_h + (size_t)bh * SPP_ * DH_;
  const _Float16* vbase = v_t + (size_t)bh * DH_ * SPP_;

  v16h aq[4];
#pragma unroll
  for (int kk = 0; kk < 4; ++kk) aq[kk] = load_a(qbase + kk * 32, DH_);

  v8f co[8] = {};
  float mrow[8], lrow[8];
#pragma unroll
  for (int r = 0; r < 8; ++r) { mrow[r] = -3.0e38f; lrow[r] = 0.f; }

  const int rbase = (lane >> 4) << 3;
  const int nl    = lane & 15;
  const int nblocks = (P_ + m0 + 16 + 31) >> 5;
  _Float16* pp = &lds_p[wslot][0];

  for (int kb = 0; kb < nblocks; ++kb) {
    const int n0 = kb * 32;
    v8f s0 = {}, s1 = {};
#pragma unroll
    for (int kk = 0; kk < 4; ++kk) {
      const v16h b0 = load_b(kbase + (size_t)n0 * DH_ + kk * 32, DH_);
      const v16h b1 = load_b(kbase + (size_t)(n0 + 16) * DH_ + kk * 32, DH_);
      s0 = wmma_f16(aq[kk], b0, s0);
      s1 = wmma_f16(aq[kk], b1, s1);
    }
#pragma unroll
    for (int r = 0; r < 8; ++r) {
      const int row = r + rbase;
      const int iq  = m0 + row;
      const int j0  = n0 + nl;
      const int j1  = j0 + 16;
      float e0 = s0[r], e1 = s1[r];
      if (j0 >= P_ && (j0 - P_) > iq) e0 = -3.0e38f;   // causal mask
      if (j1 >= P_ && (j1 - P_) > iq) e1 = -3.0e38f;
      float mx = fmaxf(e0, e1);
#pragma unroll
      for (int off = 1; off < 16; off <<= 1)
        mx = fmaxf(mx, __shfl_xor(mx, off, 32));       // row max (16-lane group)
      const float mnew = fmaxf(mrow[r], mx);
      const float corr = __expf(mrow[r] - mnew);
      const float p0 = __expf(e0 - mnew);
      const float p1 = __expf(e1 - mnew);
      float rsum = p0 + p1;
#pragma unroll
      for (int off = 1; off < 16; off <<= 1)
        rsum += __shfl_xor(rsum, off, 32);             // row sum
      lrow[r] = lrow[r] * corr + rsum;
      mrow[r] = mnew;
#pragma unroll
      for (int j = 0; j < 8; ++j) co[j][r] *= corr;    // rescale O accumulators
      pp[row * 32 + nl]      = (_Float16)p0;           // stage probs in LDS
      pp[row * 32 + 16 + nl] = (_Float16)p1;
    }
    __asm__ volatile("s_wait_dscnt 0" ::: "memory");   // LDS store->load fence
    const v16h ap = load_a(pp, 32);
#pragma unroll
    for (int j = 0; j < 8; ++j) {
      const v16h bv = load_b(vbase + (size_t)(j * 16) * SPP_ + n0, SPP_);
      co[j] = wmma_f16(ap, bv, co[j]);
    }
  }

  const int b = bh >> 4, h = bh & 15;
#pragma unroll
  for (int j = 0; j < 8; ++j) {
#pragma unroll
    for (int r = 0; r < 8; ++r) {
      const int row = r + rbase;
      const size_t tok = (size_t)b * S_ + m0 + row;
      ao[tok * HD_ + h * DH_ + j * 16 + nl] = (_Float16)(co[j][r] / lrow[r]);
    }
  }
}

// ---------------------------------------------------------------------------
// Kernel 7: output GEMM + bias -> fp32 (TDM-staged A, 32x512 per workgroup)
// ---------------------------------------------------------------------------
__global__ void out_gemm_kernel(const _Float16* __restrict__ ao,
                                const _Float16* __restrict__ wT,   // [D_, HD_]
                                const float* __restrict__ bias,
                                float* __restrict__ out) {
  __shared__ _Float16 lds_a[2 * 32 * 512];
  const int lane  = threadIdx.x & 31;
  const int wslot = threadIdx.x >> 5;
  const int cb = blockIdx.x % (D_ / 512);
  const int rb = blockIdx.x / (D_ / 512);
  const int m0 = rb * 32;
  const int n0 = cb * 512 + wslot * 64;

  v8f acc[2][4] = {};
  gemm32x64_core<HD_>(ao + (size_t)m0 * HD_, wT + (size_t)n0 * HD_, lds_a, acc);

  const int rbase = (lane >> 4) << 3;
  const int nl    = lane & 15;
#pragma unroll
  for (int t = 0; t < 2; ++t) {
#pragma unroll
    for (int j = 0; j < 4; ++j) {
      const int c = n0 + j * 16 + nl;
#pragma unroll
      for (int r = 0; r < 8; ++r) {
        const int token = m0 + t * 16 + r + rbase;
        out[(size_t)token * D_ + c] = acc[t][j][r] + bias[c];
      }
    }
  }
}

// ---------------------------------------------------------------------------
// Launcher
// ---------------------------------------------------------------------------
extern "C" void kernel_launch(void* const* d_in, const int* in_sizes, int n_in,
                              void* d_out, int out_size, void* d_ws, size_t ws_size,
                              hipStream_t stream) {
  const float* x     = (const float*)d_in[0];   // [B,S,D]
  const float* vres  = (const float*)d_in[1];   // [B,H,S,DH]
  const float* ln_g  = (const float*)d_in[2];   // [D]
  const float* ln_b  = (const float*)d_in[3];   // [D]
  const float* w_qkv = (const float*)d_in[4];   // [D, 3*H*DH]
  const float* w_out = (const float*)d_in[5];   // [H*DH, D]
  const float* b_out = (const float*)d_in[6];   // [D]
  const float* w_mix = (const float*)d_in[7];   // [D, H]
  const float* b_mix = (const float*)d_in[8];   // [H]
  const float* pmem  = (const float*)d_in[9];   // [2,H,P,DH]
  float* out = (float*)d_out;

  char* ws = (char*)d_ws;
  size_t o = 0;
  _Float16* xn    = (_Float16*)(ws + o); o += (size_t)NTOK_ * D_ * 2;
  _Float16* wqkvT = (_Float16*)(ws + o); o += (size_t)QKVN_ * D_ * 2;
  _Float16* woutT = (_Float16*)(ws + o); o += (size_t)D_ * HD_ * 2;
  float*    mixf  = (float*)   (ws + o); o += (size_t)NTOK_ * H_ * 4;
  _Float16* q_h   = (_Float16*)(ws + o); o += (size_t)B_ * H_ * S_ * DH_ * 2;
  _Float16* k_h   = (_Float16*)(ws + o); o += (size_t)B_ * H_ * SPP_ * DH_ * 2;
  _Float16* v_t   = (_Float16*)(ws + o); o += (size_t)B_ * H_ * SPP_ * DH_ * 2;
  _Float16* ao    = (_Float16*)(ws + o); o += (size_t)NTOK_ * HD_ * 2;
  (void)ws_size; (void)in_sizes; (void)n_in; (void)out_size;

  // 1) weight transpose+convert (fp32 -> f16, N-major for B-fragments)
  tconv_kernel<<<2048, 256, 0, stream>>>(w_qkv, wqkvT, D_, QKVN_);
  tconv_kernel<<<2048, 256, 0, stream>>>(w_out, woutT, HD_, D_);

  // 2) layernorm -> f16 activations
  ln_kernel<<<NTOK_, 256, 0, stream>>>(x, ln_g, ln_b, xn);

  // 3) mix gate: one wave per (token, head)
  mix_kernel<<<(NTOK_ * H_) / 8, 256, 0, stream>>>(xn, w_mix, b_mix, mixf);

  // 4) persistent-memory tokens + key-pad zeroing
  pmem_kernel<<<(H_ * P_ * DH_ + 255) / 256, 256, 0, stream>>>(pmem, k_h, v_t);
  pad_kernel<<<(B_ * H_ * (SPP_ - SP_) * DH_ + 255) / 256, 256, 0, stream>>>(k_h, v_t);

  // 5) QKV GEMM: (4096/32)*(6144/512) = 1536 workgroups of 8 waves
  qkv_gemm_kernel<<<(NTOK_ / 32) * (QKVN_ / 512), 256, 0, stream>>>(
      xn, wqkvT, mixf, vres, q_h, k_h, v_t);

  // 6) flash attention: B*H*(S/16) = 4096 waves, 8 waves/block
  attn_kernel<<<4096 / 8, 256, 0, stream>>>(q_h, k_h, v_t, ao);

  // 7) output GEMM: (4096/32)*(2048/512) = 512 workgroups
  out_gemm_kernel<<<(NTOK_ / 32) * (D_ / 512), 256, 0, stream>>>(
      ao, woutT, b_out, out);
}
